// CausalSelfAttention_57526791963252
// MI455X (gfx1250) — matrix operand order repaired and checked by reading, not compiled, not run on hardware
//
#include <hip/hip_runtime.h>

typedef __bf16 bf16;
typedef __attribute__((ext_vector_type(16))) __bf16 v16bf;
typedef __attribute__((ext_vector_type(8)))  __bf16 v8bf;
typedef __attribute__((ext_vector_type(8)))  float  v8f;
typedef __attribute__((ext_vector_type(4)))  unsigned int u32x4;
typedef __attribute__((ext_vector_type(8)))  int i32x8;
typedef __attribute__((ext_vector_type(4)))  int i32x4;

#define C_DIM   2048
#define C_HEADS 16
#define C_KVH   4
#define C_HD    128
#define C_KVDIM 512
#define C_RANK  32
#define C_B     4
#define C_S     2048
#define C_M     (C_B * C_S)

// ---------------------------------------------------------------------------
// WMMA wrapper: D = A(16x32 bf16) * B(32x16 bf16) + C(16x16 f32)
// ---------------------------------------------------------------------------
__device__ __forceinline__ v8f wmma_bf16(v16bf a, v16bf b, v8f c) {
  return __builtin_amdgcn_wmma_f32_16x16x32_bf16(
      /*neg_a=*/false, a, /*neg_b=*/false, b,
      /*c_mod=*/(short)0, c, /*reuse_a=*/false, /*reuse_b=*/false);
}

union FragU {
  v16bf full;
  struct { v8bf lo, hi; } h;
};

// A-operand (16x32, M x K), row-major source with leading dim `ld` (elements).
// Lanes 0-15: rows M=0..15, K=0..15; lanes 16-31: same rows,
// K=16..31.
__device__ __forceinline__ v16bf load_fragA(const bf16* base, int ld, int row0, int k0) {
  int lane = threadIdx.x & 31;
  int r    = lane & 15;
  int kOff = (lane >> 4) << 4;
  const bf16* p = base + (size_t)(row0 + r) * ld + k0 + kOff;
  FragU f;
  f.h.lo = *(const v8bf*)(p);
  f.h.hi = *(const v8bf*)(p + 8);
  return f.full;
}

// B-operand (32x16, K x N) supplied as B^T rows (N x K row-major).
// Lane n=lane&15 holds column n; lanes 0-15 K=0..15, lanes 16-31 K=16..31.
__device__ __forceinline__ v16bf load_fragB(const bf16* base, int ld, int n0, int k0) {
  int lane = threadIdx.x & 31;
  int n    = lane & 15;
  int kOff = (lane >> 4) << 4;
  const bf16* p = base + (size_t)(n0 + n) * ld + k0 + kOff;
  return *(const v16bf*)(p);
}

// ---------------------------------------------------------------------------
// TDM: 2-D tile load (global -> LDS), bf16 elements, row-major tensor.
// Builds D# groups per CDNA5 ISA 8.3/8.4: count=1, type=2(image),
// data_size=1 (2 bytes), 2-D tile (extra groups zero / unused).
// LDS layout produced: tile_y rows of tile_x elements, contiguous.
// ---------------------------------------------------------------------------
__device__ __forceinline__ void tdm_load_tile_bf16(
    const bf16* gsrc, unsigned lds_byte_off,
    unsigned tile_x, unsigned tile_y,
    unsigned tensor_x, unsigned tensor_y, unsigned stride_x) {
  unsigned long long ga = (unsigned long long)(const void*)gsrc;
  u32x4 g0;
  g0.x = 1u;                                              // count=1
  g0.y = lds_byte_off;                                    // lds_addr [63:32]
  g0.z = (unsigned)(ga & 0xffffffffu);                    // global_addr[31:0]
  g0.w = (unsigned)((ga >> 32) & 0x1ffffffu) | (2u << 30);// addr[56:32] | type=2
  i32x8 g1;
  g1.s0 = (int)(1u << 16);                                // data_size=1 (2B), wg_mask=0
  g1.s1 = (int)((tensor_x & 0xffffu) << 16);              // tensor_dim0[15:0]
  g1.s2 = (int)(((tensor_x >> 16) & 0xffffu) | ((tensor_y & 0xffffu) << 16));
  g1.s3 = (int)(((tensor_y >> 16) & 0xffffu) | ((tile_x & 0xffffu) << 16));
  g1.s4 = (int)(tile_y & 0xffffu);                        // tile_dim1 (tile_dim2=0)
  g1.s5 = (int)stride_x;                                  // tensor_dim0_stride[31:0]
  g1.s6 = 0;                                              // stride0[47:32]|stride1[15:0]
  g1.s7 = 0;
  i32x4 z4 = {};
  i32x8 z8 = {};
  __builtin_amdgcn_tensor_load_to_lds(g0, g1, z4, z4, z8, 0);
}

__device__ __forceinline__ unsigned lds_off_of(const void* p) {
  // generic LDS pointer: low 32 bits are the LDS byte address (flat->LDS map)
  return (unsigned)(unsigned long long)p;
}

// ---------------------------------------------------------------------------
// K1: fold LoRA into dense weight, f32 -> bf16.
// ---------------------------------------------------------------------------
__global__ void fuse_weight_kernel(const float* __restrict__ W,
                                   const float* __restrict__ Bm,
                                   const float* __restrict__ Am,
                                   bf16* __restrict__ out, int OUT, int IN) {
  int idx = blockIdx.x * blockDim.x + threadIdx.x;
  if (idx >= OUT * IN) return;
  int o = idx / IN;
  int i = idx - o * IN;
  float acc = W[idx];
#pragma unroll 8
  for (int r = 0; r < C_RANK; ++r)
    acc += Bm[o * C_RANK + r] * Am[r * IN + i];
  out[idx] = (bf16)acc;
}

// ---------------------------------------------------------------------------
// K2: f32 -> bf16 convert
// ---------------------------------------------------------------------------
__global__ void f32_to_bf16_kernel(const float* __restrict__ in, bf16* __restrict__ out, int n) {
  int idx = blockIdx.x * blockDim.x + threadIdx.x;
  if (idx < n) out[idx] = (bf16)in[idx];
}

// ---------------------------------------------------------------------------
// K3: GEMM  C[M,N] f32 = A[M,K] bf16 x W[N,K]^T bf16
// Block = 256 threads (8 waves), block tile 128x128, K-slice 32.
// TDM double-buffers A(128x32) and B(128x32) tiles into LDS; wave 0 issues
// tensor_load_to_lds, everyone syncs on s_wait_tensorcnt + barrier.
// Each wave computes a 32x64 sub-tile from LDS fragments.
// ---------------------------------------------------------------------------
#define GT_K 32
#define GT_TILE (128 * GT_K)   // elements per A (or B) tile
__global__ void gemm_xwT_tdm_kernel(const bf16* __restrict__ A, const bf16* __restrict__ W,
                                    float* __restrict__ C, int M, int N, int K) {
  // LDS: 2 buffers x (A 128x32 + B 128x32) bf16 = 32 KB
  // buffer i: A tile at smem + i*2*GT_TILE, B tile at smem + i*2*GT_TILE + GT_TILE
  __shared__ __align__(32) bf16 smem[2 * 2 * GT_TILE];

  int nBlocks = N >> 7;
  int mi = blockIdx.x / nBlocks;
  int ni = blockIdx.x - mi * nBlocks;
  int m0 = mi << 7;
  int n0 = ni << 7;

  int widb = threadIdx.x >> 5;   // 0..7
  int lane = threadIdx.x & 31;
  int wm   = widb >> 1;          // 0..3 -> 32-row slice
  int wn   = widb & 1;           // 0..1 -> 64-col slice
  int rm   = wm << 5;
  int rn   = wn << 6;

  int nIter = K / GT_K;

  if (widb == 0) {
    tdm_load_tile_bf16(A + (size_t)m0 * K, lds_off_of(smem),           GT_K, 128, K, M, K);
    tdm_load_tile_bf16(W + (size_t)n0 * K, lds_off_of(smem + GT_TILE), GT_K, 128, K, N, K);
    __builtin_amdgcn_s_wait_tensorcnt(0);
  }
  __syncthreads();

  v8f acc[2][4] = {};
  for (int it = 0; it < nIter; ++it) {
    int cur = it & 1;
    int nxt = cur ^ 1;
    if (widb == 0 && (it + 1) < nIter) {
      int k = (it + 1) * GT_K;
      bf16* An = smem + nxt * (2 * GT_TILE);
      tdm_load_tile_bf16(A + (size_t)m0 * K + k, lds_off_of(An),           GT_K, 128, K, M, K);
      tdm_load_tile_bf16(W + (size_t)n0 * K + k, lds_off_of(An + GT_TILE), GT_K, 128, K, N, K);
    }

    const bf16* Ac = smem + cur * (2 * GT_TILE);
    const bf16* Bc = Ac + GT_TILE;
    v16bf a0 = load_fragA(Ac, GT_K, rm,      0);
    v16bf a1 = load_fragA(Ac, GT_K, rm + 16, 0);
#pragma unroll
    for (int t = 0; t < 4; ++t) {
      v16bf b = load_fragB(Bc, GT_K, rn + (t << 4), 0);
      acc[0][t] = wmma_bf16(a0, b, acc[0][t]);
      acc[1][t] = wmma_bf16(a1, b, acc[1][t]);
    }

    __syncthreads();                       // all reads of `cur` done
    if (widb == 0) __builtin_amdgcn_s_wait_tensorcnt(0);
    __syncthreads();                       // `nxt` buffer now valid
  }

  int col = lane & 15, hs = lane >> 4;
#pragma unroll
  for (int i = 0; i < 2; ++i)
#pragma unroll
    for (int t = 0; t < 4; ++t)
#pragma unroll
      for (int j = 0; j < 8; ++j)
        C[(size_t)(m0 + rm + (i << 4) + j + (hs << 3)) * N + n0 + rn + (t << 4) + col] =
            acc[i][t][j];
}

// ---------------------------------------------------------------------------
// K4: per-head RMSNorm + RoPE (+ optional gain), [B,S,H,HD] f32 -> [B,H,S,HD] bf16
// ---------------------------------------------------------------------------
__global__ void norm_rope_kernel(const float* __restrict__ in, bf16* __restrict__ out,
                                 const float* __restrict__ gain, int nHeads) {
  int t = threadIdx.x;                       // 0..127
  int h = blockIdx.x % nHeads;
  int s = (blockIdx.x / nHeads) % C_S;
  int b = blockIdx.x / (nHeads * C_S);

  const float* row = in + (size_t)blockIdx.x * C_HD;
  float x = row[t];

  __shared__ float red[C_HD];
  __shared__ float sx[C_HD];
  red[t] = x * x;
  __syncthreads();
#pragma unroll
  for (int off = 64; off > 0; off >>= 1) {
    if (t < off) red[t] += red[t + off];
    __syncthreads();
  }
  float ms = red[0] * (1.0f / C_HD);
  float r  = rsqrtf(ms + 1.1920929e-7f);
  sx[t] = x * r;
  __syncthreads();

  int  i  = t & 63;
  bool hi = (t >= 64);
  float ang = (float)s * __expf(-(float)i * (9.210340371976184f / 64.0f));
  float c, sn;
  __sincosf(ang, &sn, &c);
  float x1 = sx[i], x2 = sx[i + 64];
  float o  = hi ? (x2 * c - x1 * sn) : (x1 * c + x2 * sn);
  if (gain) o *= gain[h];
  out[(((size_t)(b * nHeads + h)) * C_S + s) * C_HD + t] = (bf16)o;
}

// ---------------------------------------------------------------------------
// K5: V transpose  [B,S,KVH,HD] f32 -> [B,KVH,HD,S] bf16
// ---------------------------------------------------------------------------
__global__ void transpose_v_kernel(const float* __restrict__ vf, bf16* __restrict__ vT) {
  int idx = blockIdx.x * blockDim.x + threadIdx.x;   // over B*S*KVDIM
  int d = idx % C_HD;
  int h = (idx / C_HD) % C_KVH;
  int s = (idx / C_KVDIM) % C_S;
  int b = idx / (C_S * C_KVDIM);
  vT[(((size_t)(b * C_KVH + h)) * C_HD + d) * C_S + s] = (bf16)vf[idx];
}

// ---------------------------------------------------------------------------
// K6: flash attention with WMMA. One wave handles 16 query rows of one head.
// q: [B,H,S,HD] bf16, k: [B,KVH,S,HD] bf16, vT: [B,KVH,HD,S] bf16
// y: [B,S,H,HD] bf16
// ---------------------------------------------------------------------------
__global__ void flash_attn_kernel(const bf16* __restrict__ qn, const bf16* __restrict__ kn,
                                  const bf16* __restrict__ vT, bf16* __restrict__ y) {
  __shared__ __align__(32) bf16 pshare[8 * 16 * 32];
  int widb = threadIdx.x >> 5;
  bf16* plds = pshare + widb * (16 * 32);

  int unit = blockIdx.x * 8 + widb;
  int qt = unit % (C_S / 16);
  int bh = unit / (C_S / 16);
  int h  = bh % C_HEADS;
  int b  = bh / C_HEADS;
  int kvh = h / (C_HEADS / C_KVH);

  const bf16* qbase = qn + (size_t)(b * C_HEADS + h)  * C_S * C_HD;
  const bf16* kbase = kn + (size_t)(b * C_KVH  + kvh) * C_S * C_HD;
  const bf16* vbase = vT + (size_t)(b * C_KVH  + kvh) * C_HD * C_S;

  int qb   = qt << 4;
  int lane = threadIdx.x & 31;
  int col  = lane & 15;
  int hs   = lane >> 4;

  v16bf qf[4];
#pragma unroll
  for (int kd = 0; kd < 4; ++kd) qf[kd] = load_fragA(qbase, C_HD, qb, kd * 32);

  v8f acc[8] = {};
  float rm[8], rl[8];
#pragma unroll
  for (int j = 0; j < 8; ++j) { rm[j] = -1e30f; rl[j] = 0.0f; }

  const float scale = 0.08838834764831845f;  // 1/sqrt(128)

  for (int kb = 0; kb < qb + 16; kb += 32) {
    v8f sc0 = {}, sc1 = {};
#pragma unroll
    for (int kd = 0; kd < 4; ++kd) {
      v16bf bk0 = load_fragB(kbase, C_HD, kb,      kd * 32);
      v16bf bk1 = load_fragB(kbase, C_HD, kb + 16, kd * 32);
      sc0 = wmma_bf16(qf[kd], bk0, sc0);
      sc1 = wmma_bf16(qf[kd], bk1, sc1);
    }

#pragma unroll
    for (int j = 0; j < 8; ++j) {
      int m   = j + (hs << 3);
      int row = qb + m;
      float s0 = sc0[j] * scale;
      float s1 = sc1[j] * scale;
      if (kb +      col > row) s0 = -1e30f;
      if (kb + 16 + col > row) s1 = -1e30f;

      float v = fmaxf(s0, s1);
      v = fmaxf(v, __shfl_xor(v, 1, 32));
      v = fmaxf(v, __shfl_xor(v, 2, 32));
      v = fmaxf(v, __shfl_xor(v, 4, 32));
      v = fmaxf(v, __shfl_xor(v, 8, 32));
      float mnew = fmaxf(rm[j], v);

      float p0 = __expf(s0 - mnew);
      float p1 = __expf(s1 - mnew);
      float ps = p0 + p1;
      ps += __shfl_xor(ps, 1, 32);
      ps += __shfl_xor(ps, 2, 32);
      ps += __shfl_xor(ps, 4, 32);
      ps += __shfl_xor(ps, 8, 32);

      float alpha = __expf(rm[j] - mnew);
      rl[j] = rl[j] * alpha + ps;
      rm[j] = mnew;
#pragma unroll
      for (int dt = 0; dt < 8; ++dt) acc[dt][j] *= alpha;

      plds[m * 32 +      col] = (bf16)p0;
      plds[m * 32 + 16 + col] = (bf16)p1;
    }

    v16bf pf = load_fragA(plds, 32, 0, 0);
#pragma unroll
    for (int dt = 0; dt < 8; ++dt) {
      v16bf vfrag = load_fragB(vbase, C_S, dt << 4, kb);
      acc[dt] = wmma_bf16(pf, vfrag, acc[dt]);
    }
  }

#pragma unroll
  for (int dt = 0; dt < 8; ++dt) {
#pragma unroll
    for (int j = 0; j < 8; ++j) {
      int m = j + (hs << 3);
      float o = acc[dt][j] / rl[j];
      y[(((size_t)b * C_S + qb + m) * C_HEADS + h) * C_HD + (dt << 4) + col] = (bf16)o;
    }
  }
}

// ---------------------------------------------------------------------------
// Host launcher
// ---------------------------------------------------------------------------
extern "C" void kernel_launch(void* const* d_in, const int* in_sizes, int n_in,
                              void* d_out, int out_size, void* d_ws, size_t ws_size,
                              hipStream_t stream) {
  const float* x  = (const float*)d_in[0];
  const float* Wq = (const float*)d_in[1];
  const float* Wk = (const float*)d_in[2];
  const float* Wv = (const float*)d_in[3];
  const float* Wo = (const float*)d_in[4];
  const float* Aq = (const float*)d_in[5];
  const float* Bq = (const float*)d_in[6];
  const float* Ak = (const float*)d_in[7];
  const float* Bk = (const float*)d_in[8];
  const float* Av = (const float*)d_in[9];
  const float* Bv = (const float*)d_in[10];
  const float* Ao = (const float*)d_in[11];
  const float* Bo = (const float*)d_in[12];
  const float* qg = (const float*)d_in[13];

  char* ws = (char*)d_ws;
  bf16* xb  = (bf16*)ws;                               // M*DIM
  bf16* Wqf = xb  + (size_t)C_M * C_DIM;               // DIM*DIM
  bf16* Wkf = Wqf + (size_t)C_DIM * C_DIM;             // KVDIM*DIM
  bf16* Wvf = Wkf + (size_t)C_KVDIM * C_DIM;           // KVDIM*DIM
  bf16* Wof = Wvf + (size_t)C_KVDIM * C_DIM;           // DIM*DIM
  float* qf = (float*)(Wof + (size_t)C_DIM * C_DIM);   // M*DIM f32
  float* kf = qf + (size_t)C_M * C_DIM;                // M*KVDIM f32
  float* vf = kf + (size_t)C_M * C_KVDIM;              // M*KVDIM f32
  bf16* qn  = (bf16*)(vf + (size_t)C_M * C_KVDIM);     // M*DIM
  bf16* knb = qn + (size_t)C_M * C_DIM;                // M*KVDIM
  bf16* vTb = knb + (size_t)C_M * C_KVDIM;             // M*KVDIM
  bf16* yb  = (bf16*)qf;                               // reuse qf region

  // 1) fold LoRA into weights
  fuse_weight_kernel<<<(C_DIM * C_DIM + 255) / 256, 256, 0, stream>>>(Wq, Bq, Aq, Wqf, C_DIM, C_DIM);
  fuse_weight_kernel<<<(C_KVDIM * C_DIM + 255) / 256, 256, 0, stream>>>(Wk, Bk, Ak, Wkf, C_KVDIM, C_DIM);
  fuse_weight_kernel<<<(C_KVDIM * C_DIM + 255) / 256, 256, 0, stream>>>(Wv, Bv, Av, Wvf, C_KVDIM, C_DIM);
  fuse_weight_kernel<<<(C_DIM * C_DIM + 255) / 256, 256, 0, stream>>>(Wo, Bo, Ao, Wof, C_DIM, C_DIM);

  // 2) x -> bf16
  f32_to_bf16_kernel<<<(C_M * C_DIM + 255) / 256, 256, 0, stream>>>(x, xb, C_M * C_DIM);

  // 3) projections: block tile 128x128, TDM double-buffered
  gemm_xwT_tdm_kernel<<<(C_M / 128) * (C_DIM / 128), 256, 0, stream>>>(xb, Wqf, qf, C_M, C_DIM, C_DIM);
  gemm_xwT_tdm_kernel<<<(C_M / 128) * (C_KVDIM / 128), 256, 0, stream>>>(xb, Wkf, kf, C_M, C_KVDIM, C_DIM);
  gemm_xwT_tdm_kernel<<<(C_M / 128) * (C_KVDIM / 128), 256, 0, stream>>>(xb, Wvf, vf, C_M, C_KVDIM, C_DIM);

  // 4) RMSNorm + RoPE (+gain for q), head-major bf16 outputs
  norm_rope_kernel<<<C_B * C_S * C_HEADS, C_HD, 0, stream>>>(qf, qn, qg, C_HEADS);
  norm_rope_kernel<<<C_B * C_S * C_KVH, C_HD, 0, stream>>>(kf, knb, nullptr, C_KVH);

  // 5) V transpose to [B,KVH,HD,S] bf16
  transpose_v_kernel<<<(C_M * C_KVDIM) / 256, 256, 0, stream>>>(vf, vTb);

  // 6) flash attention: 8192 wave-units -> 1024 blocks x 8 waves
  flash_attn_kernel<<<(C_B * C_HEADS * (C_S / 16)) / 8, 256, 0, stream>>>(qn, knb, vTb, yb);

  // 7) output projection -> f32 d_out
  gemm_xwT_tdm_kernel<<<(C_M / 128) * (C_DIM / 128), 256, 0, stream>>>(yb, Wof, (float*)d_out, C_M, C_DIM, C_DIM);
}
